// BigramLanguageModel_59554016526709
// MI455X (gfx1250) — compile-verified, matrix operand order checked by reading, not run on hardware
//
#include <hip/hip_runtime.h>
#include <cstdint>
#include <cstddef>

// ---------------------------------------------------------------------------
// GPT forward for gfx1250 (MI455X). All GEMMs + attention use
// v_wmma_f32_16x16x32_f16 (wave32 WMMA), fp32 accumulation, f16 operands.
// GEMM A-tiles are staged with GLOBAL_LOAD_ASYNC_TO_LDS_B128 (ASYNCcnt) and
// double-buffered LDS; attention softmax reductions use DPP row_ror
// butterflies (VALU-only, no LDS round trips).
// ---------------------------------------------------------------------------

typedef _Float16 h16;
typedef __attribute__((ext_vector_type(16))) _Float16 v16h;
typedef __attribute__((ext_vector_type(8)))  float    v8f;

#define WMMA_F32_16x16x32_F16(a, b, c) \
  __builtin_amdgcn_wmma_f32_16x16x32_f16(false, (a), false, (b), (short)0, (c), false, false)

// ISA 7.12.2: 16-bit A/B operand layout. For lane half (lane>>4), packed
// element i (0..15) sits at contraction index:
//   i=0..7  -> k = half*8 + i          (16 contiguous bytes)
//   i=8..15 -> k = 16 + half*8 + (i-8) (16 contiguous bytes)
// So a full operand row is exactly two b128 loads:
union OpUnion { uint4 u[2]; v16h v; };
__device__ __forceinline__ v16h load_op_row(const h16* rowbase, int half) {
  OpUnion un;
  un.u[0] = *(const uint4*)(rowbase + half * 8);
  un.u[1] = *(const uint4*)(rowbase + 16 + half * 8);
  return un.v;
}
// Scalar-index form (for strided gathers that cannot vectorize).
__device__ __forceinline__ int wmma_k_idx(int i, int half) {
  int v = i >> 1, e = i & 1;
  return (v >> 2) * 16 + half * 8 + (v & 3) * 2 + e;
}

// 16-lane (DPP row) all-reduce via row_ror rotations: after ror 8,4,2,1 every
// lane of the row holds the reduction. Stays within each 16-lane half.
template <int CTRL>
__device__ __forceinline__ float dpp_mov_f32(float x) {
  return __int_as_float(__builtin_amdgcn_update_dpp(
      0, __float_as_int(x), CTRL, 0xf, 0xf, true));
}
__device__ __forceinline__ float row16_allmax(float x) {
  x = fmaxf(x, dpp_mov_f32<0x128>(x));  // row_ror:8
  x = fmaxf(x, dpp_mov_f32<0x124>(x));  // row_ror:4
  x = fmaxf(x, dpp_mov_f32<0x122>(x));  // row_ror:2
  x = fmaxf(x, dpp_mov_f32<0x121>(x));  // row_ror:1
  return x;
}
__device__ __forceinline__ float row16_allsum(float x) {
  x += dpp_mov_f32<0x128>(x);
  x += dpp_mov_f32<0x124>(x);
  x += dpp_mov_f32<0x122>(x);
  x += dpp_mov_f32<0x121>(x);
  return x;
}

// ---------------------------------------------------------------------------
// Embedding: h[b,t,:] = tok_emb[x[b,t],:] + pos_emb[t,:]
// ---------------------------------------------------------------------------
__global__ __launch_bounds__(256)
void gpt_embed_kernel(const int* __restrict__ x, const float* __restrict__ tok,
                      const float* __restrict__ pos, float* __restrict__ h,
                      int T, int D) {
  const int bt = blockIdx.x;
  const int tok_id = x[bt];
  const int t = bt % T;
  const float* ts = tok + (size_t)tok_id * D;
  const float* ps = pos + (size_t)t * D;
  float* hr = h + (size_t)bt * D;
  for (int i = threadIdx.x; i < D; i += 256)
    hr[i] = ts[i] + ps[i];
}

// ---------------------------------------------------------------------------
// LayerNorm -> f16 output (feeds WMMA GEMMs). One block per row.
// ---------------------------------------------------------------------------
__global__ __launch_bounds__(256)
void gpt_ln_kernel(const float* __restrict__ x, const float* __restrict__ s,
                   const float* __restrict__ b, h16* __restrict__ out, int D) {
  __shared__ float red_s[8], red_q[8];
  __shared__ float mu_sh, rstd_sh;
  const int row = blockIdx.x, tid = threadIdx.x;
  const float* xr = x + (size_t)row * D;
  float sum = 0.f, sq = 0.f;
  for (int i = tid; i < D; i += 256) { float v = xr[i]; sum += v; sq += v * v; }
  for (int m = 16; m > 0; m >>= 1) { sum += __shfl_xor(sum, m); sq += __shfl_xor(sq, m); }
  if ((tid & 31) == 0) { red_s[tid >> 5] = sum; red_q[tid >> 5] = sq; }
  __syncthreads();
  if (tid == 0) {
    float ts = 0.f, tq = 0.f;
    for (int i = 0; i < 8; ++i) { ts += red_s[i]; tq += red_q[i]; }
    float mu = ts / D;
    float var = tq / D - mu * mu;
    mu_sh = mu;
    rstd_sh = rsqrtf(var + 1e-5f);
  }
  __syncthreads();
  const float mu = mu_sh, rstd = rstd_sh;
  h16* orow = out + (size_t)row * D;
  for (int i = tid; i < D; i += 256)
    orow[i] = (h16)((xr[i] - mu) * rstd * s[i] + b[i]);
}

// ---------------------------------------------------------------------------
// GEMM: out[M,N] = act(A[M,K] @ B[K,N] + bias) (+ Cin residual)
//   A: f16 row-major (activations) -> async DMA into LDS (no VGPR staging).
//   B: f32 row-major (weights) -> registers early, f32->f16 convert + LDS
//      store after compute (latency hidden under the WMMAs).
//   Double-buffered LDS (2 x 32 KB). 256 threads = 8 waves, block tile
//   128x128; each wave owns 32x64 -> 8 v_wmma_f32_16x16x32_f16 per 32-K
//   sub-chunk, K staged in tiles of 64.
// Requires: M%128==0, N%128==0, K%64==0 (true for all call sites).
// ---------------------------------------------------------------------------
template <bool HAS_BIAS, bool RELU, bool HAS_CIN, bool HAS_COUT, bool HAS_DH>
__global__ __launch_bounds__(256)
void gpt_gemm_kernel(const h16* __restrict__ A, const float* __restrict__ Bw,
                     const float* __restrict__ bias, const float* __restrict__ Cin,
                     float* __restrict__ Cout, h16* __restrict__ Dh,
                     int M, int N, int K) {
  __shared__ h16 lds_a[2][128 * 64];   // [buf][m][k]  2 x 16 KB
  __shared__ h16 lds_bt[2][128 * 64];  // [buf][n][k]  2 x 16 KB (transposed)
  const int tid = threadIdx.x;
  const int bm = blockIdx.y * 128;
  const int bn = blockIdx.x * 128;
  const int w = tid >> 5, l = tid & 31;
  const int wm = (w >> 1) * 32;     // 0,32,64,96
  const int wn = (w & 1) * 64;      // 0,64
  const int lane16 = l & 15, half = l >> 4;

  // staging assignments
  const int ar = tid >> 1, ac = (tid & 1) * 32;        // A: 64 B per thread
  const int bncol = tid & 127, bkr = (tid >> 7) * 32;  // B: 32 k-rows/thread

  v8f acc[2][4] = {};
  float breg[32];
  const int nt = K / 64;

  // A tile: per-lane async gather into LDS (ASYNCcnt). offset: applies to
  // both the LDS and global addresses (ISA 15.18.3), so one base covers 64 B.
  auto stage_a_async = [&](int t, int buf) {
    const h16* g = A + (size_t)(bm + ar) * K + (t * 64 + ac);
    unsigned lo = (unsigned)(uintptr_t)(&lds_a[buf][ar * 64 + ac]);
    asm volatile(
        "global_load_async_to_lds_b128 %0, %1, off\n\t"
        "global_load_async_to_lds_b128 %0, %1, off offset:16\n\t"
        "global_load_async_to_lds_b128 %0, %1, off offset:32\n\t"
        "global_load_async_to_lds_b128 %0, %1, off offset:48"
        :: "v"(lo), "v"(g) : "memory");
  };
  auto wait_async = [&]() {
    asm volatile("s_wait_asynccnt 0" ::: "memory");
  };
  auto load_b_regs = [&](int t) {
    const float* bsrc = Bw + (size_t)(t * 64 + bkr) * N + (bn + bncol);
#pragma unroll
    for (int i = 0; i < 32; ++i) breg[i] = bsrc[(size_t)i * N];
  };
  auto store_b = [&](int buf) {
    h16* bdst = &lds_bt[buf][bncol * 64 + bkr];
#pragma unroll
    for (int i = 0; i < 32; ++i) bdst[i] = (h16)breg[i];
  };
  auto compute = [&](int buf) {
#pragma unroll
    for (int kc = 0; kc < 64; kc += 32) {
      const v16h a0 = load_op_row(&lds_a[buf][(wm + 0  + lane16) * 64 + kc], half);
      const v16h a1 = load_op_row(&lds_a[buf][(wm + 16 + lane16) * 64 + kc], half);
      const v16h b0 = load_op_row(&lds_bt[buf][(wn + 0  + lane16) * 64 + kc], half);
      const v16h b1 = load_op_row(&lds_bt[buf][(wn + 16 + lane16) * 64 + kc], half);
      const v16h b2 = load_op_row(&lds_bt[buf][(wn + 32 + lane16) * 64 + kc], half);
      const v16h b3 = load_op_row(&lds_bt[buf][(wn + 48 + lane16) * 64 + kc], half);
      acc[0][0] = WMMA_F32_16x16x32_F16(a0, b0, acc[0][0]);
      acc[0][1] = WMMA_F32_16x16x32_F16(a0, b1, acc[0][1]);
      acc[0][2] = WMMA_F32_16x16x32_F16(a0, b2, acc[0][2]);
      acc[0][3] = WMMA_F32_16x16x32_F16(a0, b3, acc[0][3]);
      acc[1][0] = WMMA_F32_16x16x32_F16(a1, b0, acc[1][0]);
      acc[1][1] = WMMA_F32_16x16x32_F16(a1, b1, acc[1][1]);
      acc[1][2] = WMMA_F32_16x16x32_F16(a1, b2, acc[1][2]);
      acc[1][3] = WMMA_F32_16x16x32_F16(a1, b3, acc[1][3]);
    }
  };

  // prologue: fill buffer 0
  stage_a_async(0, 0);
  load_b_regs(0);
  store_b(0);
  wait_async();
  __syncthreads();

  for (int t = 0; t < nt; ++t) {
    const int cur = t & 1, nxt = cur ^ 1;
    const bool has_next = (t + 1) < nt;
    if (has_next) {
      stage_a_async(t + 1, nxt);  // DMA next A tile (buffer freed at the
                                  // previous barrier)
      load_b_regs(t + 1);         // issue next B loads into registers
    }
    compute(cur);                 // 16 WMMAs hide the above latencies
    if (has_next) {
      store_b(nxt);
      wait_async();
    }
    __syncthreads();
  }

  // Epilogue. C/D layout (ISA 7.12.2): element (m,n) -> vgpr m%8,
  // lane (m/8)*16 + n. Straight-line (all flags compile-time).
#pragma unroll
  for (int mi = 0; mi < 2; ++mi) {
#pragma unroll
    for (int j = 0; j < 4; ++j) {
#pragma unroll
      for (int r = 0; r < 8; ++r) {
        const int m = r + half * 8;
        const int row = bm + wm + mi * 16 + m;
        const int col = bn + wn + j * 16 + lane16;
        float vv = acc[mi][j][r];
        if (HAS_BIAS) vv += bias[col];
        if (RELU)     vv = vv > 0.f ? vv : 0.f;
        const size_t idx = (size_t)row * N + col;
        if (HAS_CIN)  vv += Cin[idx];
        if (HAS_COUT) Cout[idx] = vv;
        if (HAS_DH)   Dh[idx]   = (h16)vv;
      }
    }
  }
}

// ---------------------------------------------------------------------------
// Causal flash attention: one wave per (batch, head, 16-query tile).
// Per 32-key chunk: S (16x32) via 4 WMMAs, online softmax (DPP row_ror
// all-reduce within the 16-lane halves of the C layout), P staged through LDS
// into A layout, O += P @ V via 4 WMMAs. Exactly one chunk straddles the
// causal diagonal, so masking is confined to a single compile-time phase.
// ---------------------------------------------------------------------------
__global__ __launch_bounds__(32)
void gpt_attn_kernel(const h16* __restrict__ q, const h16* __restrict__ k,
                     const h16* __restrict__ v, h16* __restrict__ o,
                     int B, int T, int H, int D) {
  const int HD = 64;
  __shared__ h16 lds_p[16 * 32];
  const int l = threadIdx.x, lane16 = l & 15, half = l >> 4;
  const int nqt = T / 16;
  const int bid = blockIdx.x;
  const int qtile = bid % nqt;
  const int hh = (bid / nqt) % H;
  const int bb = bid / (nqt * H);
  const int q0 = qtile * 16;

  // Q as WMMA A-operands (b128 loads, then packed-f16 scale by HD^-0.5)
  v16h aq0, aq1;
  {
    const h16* qrow = q + ((size_t)(bb * T + q0 + lane16) * D + hh * HD);
    aq0 = load_op_row(qrow, half);
    aq1 = load_op_row(qrow + 32, half);
    const h16 sc = (h16)0.125f;
#pragma unroll
    for (int i = 0; i < 16; ++i) { aq0[i] *= sc; aq1[i] *= sc; }
  }

  v8f acc[4] = {};
  float mi[8], li[8];
#pragma unroll
  for (int r = 0; r < 8; ++r) { mi[r] = -1e30f; li[r] = 0.f; }

  const int kfull = q0 & ~31;  // keys [0,kfull) need no causal mask

#pragma unroll
  for (int phase = 0; phase < 2; ++phase) {
    const bool MASKED = (phase == 1);
    const int kc_lo = MASKED ? kfull : 0;
    const int kc_hi = MASKED ? kfull + 32 : kfull;
    for (int kc = kc_lo; kc < kc_hi; kc += 32) {
      // S = Q @ K^T for keys [kc, kc+32)
      v8f s0 = {}, s1 = {};
      {
        const h16* kr0 = k + ((size_t)(bb * T + kc +      lane16) * D + hh * HD);
        const h16* kr1 = k + ((size_t)(bb * T + kc + 16 + lane16) * D + hh * HD);
        const v16h b00 = load_op_row(kr0, half);
        const v16h b01 = load_op_row(kr0 + 32, half);
        const v16h b10 = load_op_row(kr1, half);
        const v16h b11 = load_op_row(kr1 + 32, half);
        s0 = WMMA_F32_16x16x32_F16(aq0, b00, s0);
        s0 = WMMA_F32_16x16x32_F16(aq1, b01, s0);
        s1 = WMMA_F32_16x16x32_F16(aq0, b10, s1);
        s1 = WMMA_F32_16x16x32_F16(aq1, b11, s1);
      }

      // Online softmax update; mask only in the diagonal phase.
#pragma unroll
      for (int r = 0; r < 8; ++r) {
        const int m = r + half * 8;
        float x0 = s0[r], x1 = s1[r];
        if (MASKED) {
          const int lim = q0 + m;
          x0 = (kc + lane16      <= lim) ? x0 : -1e30f;
          x1 = (kc + 16 + lane16 <= lim) ? x1 : -1e30f;
        }
        const float rm = row16_allmax(fmaxf(x0, x1));
        const float nm = fmaxf(mi[r], rm);
        const float sc = __expf(mi[r] - nm);
        mi[r] = nm;
        const float p0 = __expf(x0 - nm);
        const float p1 = __expf(x1 - nm);
        const float rs = row16_allsum(p0 + p1);
        li[r] = li[r] * sc + rs;
        acc[0][r] *= sc; acc[1][r] *= sc; acc[2][r] *= sc; acc[3][r] *= sc;
        lds_p[m * 32 + lane16]      = (h16)p0;
        lds_p[m * 32 + 16 + lane16] = (h16)p1;
      }
      __syncthreads();

      // P (C layout) -> A-operand layout via LDS round trip
      const v16h ap = load_op_row(&lds_p[lane16 * 32], half);
      __syncthreads();

      // O += P @ V  (V chunk: 32 keys x 64 hd, 4 B-operand N-tiles; keys are
      // the contraction dim -> per-element strided gather, coalesced across
      // the 16 lanes of each half).
      const h16* vbase = v + ((size_t)(bb * T + kc) * D + hh * HD);
#pragma unroll
      for (int j = 0; j < 4; ++j) {
        v16h bv;
#pragma unroll
        for (int i = 0; i < 16; ++i) {
          const int kk = wmma_k_idx(i, half);  // key within chunk
          bv[i] = vbase[(size_t)kk * D + j * 16 + lane16];
        }
        acc[j] = WMMA_F32_16x16x32_F16(ap, bv, acc[j]);
      }
    }
  }

  // Normalize and write O (f16, feeds the wo GEMM)
#pragma unroll
  for (int r = 0; r < 8; ++r) {
    const int m = r + half * 8;
    const float inv = 1.f / li[r];
    h16* orow = o + ((size_t)(bb * T + q0 + m) * D + hh * HD);
#pragma unroll
    for (int j = 0; j < 4; ++j)
      orow[j * 16 + lane16] = (h16)(acc[j][r] * inv);
  }
}

// ---------------------------------------------------------------------------
// Host orchestration
// ---------------------------------------------------------------------------
extern "C" void kernel_launch(void* const* d_in, const int* in_sizes, int n_in,
                              void* d_out, int out_size, void* d_ws, size_t ws_size,
                              hipStream_t stream) {
  (void)in_sizes; (void)n_in; (void)out_size; (void)ws_size;
  constexpr int B = 4, T = 1024, L = 6, D = 768, H = 12, FF = 3072, V = 32000;
  const int M = B * T;  // 4096

  const int*   x       = (const int*)  d_in[0];
  const float* tok_emb = (const float*)d_in[1];
  const float* pos_emb = (const float*)d_in[2];
  const float* wq      = (const float*)d_in[3];
  const float* wk      = (const float*)d_in[4];
  const float* wv      = (const float*)d_in[5];
  const float* wo      = (const float*)d_in[6];
  const float* bo      = (const float*)d_in[7];
  const float* ln1_s   = (const float*)d_in[8];
  const float* ln1_b   = (const float*)d_in[9];
  const float* ln2_s   = (const float*)d_in[10];
  const float* ln2_b   = (const float*)d_in[11];
  const float* w1      = (const float*)d_in[12];
  const float* b1      = (const float*)d_in[13];
  const float* w2      = (const float*)d_in[14];
  const float* b2      = (const float*)d_in[15];
  const float* lnf_s   = (const float*)d_in[16];
  const float* lnf_b   = (const float*)d_in[17];
  const float* w_head  = (const float*)d_in[18];
  const float* b_head  = (const float*)d_in[19];

  // Workspace: h(f32) + f16 activations (a, q, k, v, o, ffn-mid) ~ 69 MB.
  char* p = (char*)d_ws;
  auto take = [&](size_t bytes) -> void* {
    void* r = (void*)p;
    p += (bytes + 255) & ~(size_t)255;
    return r;
  };
  float* h = (float*)take(sizeof(float) * (size_t)M * D);
  h16* a_h = (h16*)take(sizeof(h16) * (size_t)M * D);
  h16* q_h = (h16*)take(sizeof(h16) * (size_t)M * D);
  h16* k_h = (h16*)take(sizeof(h16) * (size_t)M * D);
  h16* v_h = (h16*)take(sizeof(h16) * (size_t)M * D);
  h16* o_h = (h16*)take(sizeof(h16) * (size_t)M * D);
  h16* u_h = (h16*)take(sizeof(h16) * (size_t)M * FF);

  const dim3 blk256(256);
  const dim3 gD(D / 128, M / 128);   // 6 x 32
  const dim3 gF(FF / 128, M / 128);  // 24 x 32
  const dim3 gV(V / 128, M / 128);   // 250 x 32

  gpt_embed_kernel<<<M, blk256, 0, stream>>>(x, tok_emb, pos_emb, h, T, D);

  for (int i = 0; i < L; ++i) {
    const float* wq_i = wq + (size_t)i * D * D;
    const float* wk_i = wk + (size_t)i * D * D;
    const float* wv_i = wv + (size_t)i * D * D;
    const float* wo_i = wo + (size_t)i * D * D;
    const float* w1_i = w1 + (size_t)i * D * FF;
    const float* w2_i = w2 + (size_t)i * FF * D;

    // a = LN1(h)  (f16)
    gpt_ln_kernel<<<M, blk256, 0, stream>>>(h, ln1_s + (size_t)i * D,
                                            ln1_b + (size_t)i * D, a_h, D);
    // q, k, v = a @ w{q,k,v}  (f16 outputs only)
    gpt_gemm_kernel<false, false, false, false, true><<<gD, blk256, 0, stream>>>(
        a_h, wq_i, nullptr, nullptr, nullptr, q_h, M, D, D);
    gpt_gemm_kernel<false, false, false, false, true><<<gD, blk256, 0, stream>>>(
        a_h, wk_i, nullptr, nullptr, nullptr, k_h, M, D, D);
    gpt_gemm_kernel<false, false, false, false, true><<<gD, blk256, 0, stream>>>(
        a_h, wv_i, nullptr, nullptr, nullptr, v_h, M, D, D);
    // o = causal_softmax(q k^T / sqrt(hd)) v
    gpt_attn_kernel<<<B * H * (T / 16), 32, 0, stream>>>(q_h, k_h, v_h, o_h,
                                                         B, T, H, D);
    // h = h + o @ wo + bo
    gpt_gemm_kernel<true, false, true, true, false><<<gD, blk256, 0, stream>>>(
        o_h, wo_i, bo + (size_t)i * D, h, h, nullptr, M, D, D);
    // f = LN2(h); u = relu(f @ w1 + b1); h = h + u @ w2 + b2
    gpt_ln_kernel<<<M, blk256, 0, stream>>>(h, ln2_s + (size_t)i * D,
                                            ln2_b + (size_t)i * D, a_h, D);
    gpt_gemm_kernel<true, true, false, false, true><<<gF, blk256, 0, stream>>>(
        a_h, w1_i, b1 + (size_t)i * FF, nullptr, nullptr, u_h, M, FF, D);
    gpt_gemm_kernel<true, false, true, true, false><<<gD, blk256, 0, stream>>>(
        u_h, w2_i, b2 + (size_t)i * D, h, h, nullptr, M, D, FF);
  }

  // logits = LN_f(h) @ w_head + b_head  -> d_out (fp32)
  gpt_ln_kernel<<<M, blk256, 0, stream>>>(h, lnf_s, lnf_b, a_h, D);
  gpt_gemm_kernel<true, false, false, true, false><<<gV, blk256, 0, stream>>>(
      a_h, w_head, b_head, nullptr, (float*)d_out, nullptr, M, V, D);
}
